// LightStereoNet_63840393888418
// MI455X (gfx1250) — matrix operand order repaired and checked by reading, not compiled, not run on hardware
//
#include <hip/hip_runtime.h>
#include <hip/hip_bf16.h>

// LightStereoNet on MI455X (gfx1250, wave32).
// Heavy convs: implicit GEMM on v_wmma_f32_16x16x32_f16, NHWC activations:
//   K = tap*CIN + c (CIN in {8,16,32}) -> A fragment = 2 contiguous 16B runs
//   (global_load_b128 x2); weights staged transposed in LDS -> B fragment =
//   2 contiguous 16B runs (ds_load_b128 x2). Bounds handled by clamp+select
//   (no exec branches); all indices 32-bit (buffers < 2^31 bytes).
// Tiny-K layers (CIN*TAPS<=32): single K-chunk; fragment build is templated
// on the lane half so all tap decodes are compile-time constants.
// Fragment layouts per CDNA5 ISA 7.12.2 (wave32).
// Workspace requirement ~137 MB.

typedef __attribute__((ext_vector_type(16))) _Float16 v16h;
typedef __attribute__((ext_vector_type(8)))  _Float16 v8h;
typedef __attribute__((ext_vector_type(8)))  float    v8f;
typedef _Float16 half_t;

__device__ __forceinline__ int iclamp(int v, int lo, int hi) {
  return v < lo ? lo : (v > hi ? hi : v);
}

// ---------------------------------------------------------------------------
// A-fragment 8-wide gathers (NHWC, clamp+select, 32-bit indexing)
// ---------------------------------------------------------------------------
template<int CIN>
__device__ __forceinline__ v8h gatherA8_2d(const half_t* __restrict__ bin,
                                           int k, int HIN, int WIN,
                                           int yin0, int xin0) {
  const int tap = k / CIN;                  // CIN power of 2 -> shift
  const int c0  = k - tap * CIN;
  const int ky  = tap / 3;
  const int kx  = tap - ky * 3;
  const int yy  = yin0 + ky;
  const int xx  = xin0 + kx;
  const bool inb = (tap < 9) & (yy >= 0) & (yy < HIN) & (xx >= 0) & (xx < WIN);
  const int yyc = iclamp(yy, 0, HIN - 1);
  const int xxc = iclamp(xx, 0, WIN - 1);
  const v8h val = *(const v8h*)(bin + (yyc * WIN + xxc) * CIN + c0);
  const v8h z = {};
  return inb ? val : z;
}

template<int CIN>
__device__ __forceinline__ v8h gatherA8_3d(const half_t* __restrict__ bin,
                                           int k, int DD, int HH, int WW,
                                           int zd, int y, int xin0) {
  const int tap = k / CIN;
  const int c0  = k - tap * CIN;
  const int kd  = tap / 9;
  const int t   = tap - kd * 9;
  const int ky  = t / 3;
  const int kx  = t - ky * 3;
  const int zz  = zd + kd - 1;
  const int yy  = y  + ky - 1;
  const int xx  = xin0 + kx;
  const bool inb = (tap < 27) & (zz >= 0) & (zz < DD) & (yy >= 0) & (yy < HH) &
                   (xx >= 0) & (xx < WW);
  const int zzc = iclamp(zz, 0, DD - 1);
  const int yyc = iclamp(yy, 0, HH - 1);
  const int xxc = iclamp(xx, 0, WW - 1);
  const v8h val = *(const v8h*)(bin + ((zzc * HH + yyc) * WW + xxc) * CIN + c0);
  const v8h z = {};
  return inb ? val : z;
}

// OMODE: 0 = f16 NHWC only, 1 = f16 NHWC + f32 NCHW, 2 = f32 only
// ---------------------------------------------------------------------------
// Heavy conv2d 3x3 (NHWC), WMMA, LDS transposed weights.
// ---------------------------------------------------------------------------
template<int CIN, int COUT, int STRIDE, bool HAS_BN, bool HAS_RELU, int OMODE>
__global__ void __launch_bounds__(256)
conv2d_nhwc_wmma(const half_t* __restrict__ in, const float* __restrict__ w,
                 const float* __restrict__ scale, const float* __restrict__ bias,
                 half_t* __restrict__ out16, float* __restrict__ out32,
                 int NB, int HIN, int WIN, int HOUT, int WOUT) {
  constexpr int K     = CIN * 9;
  constexpr int KPAD  = ((K + 31) / 32) * 32;
  constexpr int KCH   = KPAD / 32;
  constexpr int NT    = (COUT + 15) / 16;
  constexpr int COUTP = NT * 16;

  __shared__ half_t wT[COUTP][KPAD];
  for (int idx = threadIdx.x; idx < COUTP * KPAD; idx += blockDim.x) {
    const int co = idx / KPAD;
    const int k  = idx - co * KPAD;
    half_t v = (half_t)0.f;
    if (co < COUT && k < K) {
      const int tap = k / CIN;
      const int c   = k - tap * CIN;
      v = (half_t)w[(co * CIN + c) * 9 + tap];
    }
    wT[co][k] = v;
  }
  __syncthreads();

  const int WT     = WOUT >> 4;
  const int tilesB = NT * HOUT * WT;
  const int total  = tilesB * NB;
  const int lane   = threadIdx.x & 31;
  int tile = blockIdx.x * (blockDim.x >> 5) + (threadIdx.x >> 5);
  const bool active = tile < total;
  if (!active) tile = total - 1;

  const int b  = tile / tilesB;
  int rem      = tile - b * tilesB;
  const int nt = rem / (HOUT * WT);  rem -= nt * HOUT * WT;
  const int y  = rem / WT;
  const int x0 = (rem - y * WT) << 4;

  const int m    = lane & 15;
  const int half = lane >> 4;
  const int co   = nt * 16 + m;
  const int yin0 = y * STRIDE - 1;
  const int xin0 = (x0 + m) * STRIDE - 1;
  const half_t* bin = in + b * (HIN * WIN * CIN);

  v8f acc = {};
#pragma unroll
  for (int ch = 0; ch < KCH; ++ch) {
    const int kc = ch * 32;
    const v8h lo = gatherA8_2d<CIN>(bin, kc + half * 8,      HIN, WIN, yin0, xin0);
    const v8h hi = gatherA8_2d<CIN>(bin, kc + 16 + half * 8, HIN, WIN, yin0, xin0);
    const v16h a = __builtin_shufflevector(lo, hi, 0,1,2,3,4,5,6,7,8,9,10,11,12,13,14,15);
    const half_t* wr = &wT[co][kc + half * 8];
    const v8h b0 = *(const v8h*)wr;
    const v8h b1 = *(const v8h*)(wr + 16);
    const v16h bf = __builtin_shufflevector(b0, b1, 0,1,2,3,4,5,6,7,8,9,10,11,12,13,14,15);
    acc = __builtin_amdgcn_wmma_f32_16x16x32_f16(false, a, false, bf,
                                                 (short)0, acc, false, false);
  }

  if (active && co < COUT) {
    const float sc = HAS_BN ? scale[co] : 1.f;
    const float bi = HAS_BN ? bias[co]  : 0.f;
    float vals[8];
#pragma unroll
    for (int r = 0; r < 8; ++r) {
      float v = acc[r];
      if (HAS_BN)   v = v * sc + bi;
      if (HAS_RELU) v = fmaxf(v, 0.f);
      vals[r] = v;
    }
    if (OMODE != 2) {
      const int rowBase = (b * HOUT + y) * WOUT + x0 + half * 8;
#pragma unroll
      for (int r = 0; r < 8; ++r)
        out16[(rowBase + r) * COUT + co] = (half_t)vals[r];
    }
    if (OMODE != 0) {  // NCHW f32 (fl / disp)
      const int chBase = ((b * COUT + co) * HOUT + y) * WOUT + x0 + half * 8;
#pragma unroll
      for (int r = 0; r < 8; ++r)
        out32[chBase + r] = vals[r];
    }
  }
}

// ---------------------------------------------------------------------------
// Heavy conv3d 3x3x3 (NHWC, per-batch pointers), WMMA + LDS weights.
// ---------------------------------------------------------------------------
template<int CIN, int COUT, bool HAS_BN, bool HAS_RELU, int OMODE>
__global__ void __launch_bounds__(256)
conv3d_nhwc_wmma(const half_t* __restrict__ in, const float* __restrict__ w,
                 const float* __restrict__ scale, const float* __restrict__ bias,
                 half_t* __restrict__ out16, float* __restrict__ out32,
                 int DD, int HH, int WW) {
  constexpr int K     = CIN * 27;
  constexpr int KPAD  = ((K + 31) / 32) * 32;
  constexpr int KCH   = KPAD / 32;
  constexpr int NT    = (COUT + 15) / 16;
  constexpr int COUTP = NT * 16;

  __shared__ half_t wT[COUTP][KPAD];
  for (int idx = threadIdx.x; idx < COUTP * KPAD; idx += blockDim.x) {
    const int co = idx / KPAD;
    const int k  = idx - co * KPAD;
    half_t v = (half_t)0.f;
    if (co < COUT && k < K) {
      const int tap = k / CIN;
      const int c   = k - tap * CIN;
      v = (half_t)w[(co * CIN + c) * 27 + tap];
    }
    wT[co][k] = v;
  }
  __syncthreads();

  const int WT    = WW >> 4;
  const int total = NT * DD * HH * WT;
  const int lane  = threadIdx.x & 31;
  int tile = blockIdx.x * (blockDim.x >> 5) + (threadIdx.x >> 5);
  const bool active = tile < total;
  if (!active) tile = total - 1;

  const int nt = tile / (DD * HH * WT);
  int rem      = tile - nt * DD * HH * WT;
  const int zd = rem / (HH * WT);    rem -= zd * HH * WT;
  const int y  = rem / WT;
  const int x0 = (rem - y * WT) << 4;

  const int m    = lane & 15;
  const int half = lane >> 4;
  const int co   = nt * 16 + m;
  const int xin0 = x0 + m - 1;

  v8f acc = {};
#pragma unroll
  for (int ch = 0; ch < KCH; ++ch) {
    const int kc = ch * 32;
    const v8h lo = gatherA8_3d<CIN>(in, kc + half * 8,      DD, HH, WW, zd, y, xin0);
    const v8h hi = gatherA8_3d<CIN>(in, kc + 16 + half * 8, DD, HH, WW, zd, y, xin0);
    const v16h a = __builtin_shufflevector(lo, hi, 0,1,2,3,4,5,6,7,8,9,10,11,12,13,14,15);
    const half_t* wr = &wT[co][kc + half * 8];
    const v8h b0 = *(const v8h*)wr;
    const v8h b1 = *(const v8h*)(wr + 16);
    const v16h bf = __builtin_shufflevector(b0, b1, 0,1,2,3,4,5,6,7,8,9,10,11,12,13,14,15);
    acc = __builtin_amdgcn_wmma_f32_16x16x32_f16(false, a, false, bf,
                                                 (short)0, acc, false, false);
  }

  if (active && co < COUT) {
    const float sc = HAS_BN ? scale[co] : 1.f;
    const float bi = HAS_BN ? bias[co]  : 0.f;
    float vals[8];
#pragma unroll
    for (int r = 0; r < 8; ++r) {
      float v = acc[r];
      if (HAS_BN)   v = v * sc + bi;
      if (HAS_RELU) v = fmaxf(v, 0.f);
      vals[r] = v;
    }
    const int pixBase = (zd * HH + y) * WW + x0 + half * 8;
    if (OMODE != 2) {
#pragma unroll
      for (int r = 0; r < 8; ++r)
        out16[(pixBase + r) * COUT + co] = (half_t)vals[r];
    }
    if (OMODE != 0) {  // COUT==1: f32 voxel grid
#pragma unroll
      for (int r = 0; r < 8; ++r)
        out32[pixBase + r] = vals[r];
    }
  }
}

// ---------------------------------------------------------------------------
// Tiny-K conv fragment build: HALF templated -> all tap decodes constant.
// K ordered c*TAPS+tap. Input NCHW (per-batch base), TIN f32 or f16.
// ---------------------------------------------------------------------------
template<typename TIN, int CIN, int COUT, int TAPS, bool IS3D, int HALF>
__device__ __forceinline__ void smallk_frags(const TIN* __restrict__ bin,
                                             const float* __restrict__ w,
                                             int co, int DD, int HIN, int WIN,
                                             int zd, int yin0, int xin0,
                                             v16h& a, v16h& bf) {
  constexpr int K = CIN * TAPS;
  const int coc = iclamp(co, 0, COUT - 1);
#pragma unroll
  for (int v = 0; v < 8; ++v) {
#pragma unroll
    for (int e = 0; e < 2; ++e) {
      // ---- A element (constant decode) ----
      {
        const int kk = ((v < 4) ? (HALF * 8 + 2 * v) : (16 + HALF * 8 + 2 * (v - 4))) + e;
        half_t av = (half_t)0.f;
        if (kk < K) {
          const int c   = kk / TAPS;
          const int tap = kk - c * TAPS;
          const int kd  = IS3D ? (tap / 9) : 0;
          const int t   = IS3D ? (tap - kd * 9) : tap;
          const int ky  = t / 3;
          const int kx  = t - ky * 3;
          const int zz  = zd + kd - (IS3D ? 1 : 0);
          const int yy  = yin0 + ky;
          const int xx  = xin0 + kx;
          const bool inb = (!IS3D || (zz >= 0 && zz < DD)) &&
                           yy >= 0 && yy < HIN && xx >= 0 && xx < WIN;
          const int zzc = IS3D ? iclamp(zz, 0, DD - 1) : 0;
          const int yyc = iclamp(yy, 0, HIN - 1);
          const int xxc = iclamp(xx, 0, WIN - 1);
          const half_t raw = (half_t)bin[((c * DD + zzc) * HIN + yyc) * WIN + xxc];
          av = inb ? raw : (half_t)0.f;
        }
        a[2 * v + e] = av;
      }
      // ---- B element (constant decode; co clamped, masked) ----
      {
        const int kk = ((v < 4) ? (HALF * 8 + 2 * v) : (16 + HALF * 8 + 2 * (v - 4))) + e;
        half_t bv = (half_t)0.f;
        if (kk < K) {
          const int c   = kk / TAPS;
          const int tap = kk - c * TAPS;
          const half_t raw = (half_t)w[(coc * CIN + c) * TAPS + tap];
          bv = (co < COUT) ? raw : (half_t)0.f;
        }
        bf[2 * v + e] = bv;
      }
    }
  }
}

template<typename TIN, int CIN, int COUT, int TAPS, int STRIDE, bool IS3D,
         bool HAS_BN, bool HAS_RELU>
__global__ void __launch_bounds__(256)
conv_smallk_wmma(const TIN* __restrict__ in, const float* __restrict__ w,
                 const float* __restrict__ scale, const float* __restrict__ bias,
                 half_t* __restrict__ out16,
                 int NB, int DD, int HIN, int WIN, int HOUT, int WOUT) {
  constexpr int K  = CIN * TAPS;
  constexpr int NT = (COUT + 15) / 16;
  static_assert(K <= 32, "single chunk only");

  const int WT     = WOUT >> 4;
  const int tilesB = NT * DD * HOUT * WT;
  const int total  = tilesB * NB;
  const int lane   = threadIdx.x & 31;
  int tile = blockIdx.x * (blockDim.x >> 5) + (threadIdx.x >> 5);
  const bool active = tile < total;
  if (!active) tile = total - 1;

  const int b  = tile / tilesB;
  int rem      = tile - b * tilesB;
  const int nt = rem / (DD * HOUT * WT);  rem -= nt * DD * HOUT * WT;
  const int zd = rem / (HOUT * WT);       rem -= zd * HOUT * WT;
  const int y  = rem / WT;
  const int x0 = (rem - y * WT) << 4;

  const int m    = lane & 15;
  const int half = lane >> 4;
  const int co   = nt * 16 + m;
  const int yin0 = y * STRIDE - 1;
  const int xin0 = (x0 + m) * STRIDE - 1;
  const TIN* bin = in + b * (CIN * DD * HIN * WIN);

  v16h a, bf;
  if (half == 0)
    smallk_frags<TIN, CIN, COUT, TAPS, IS3D, 0>(bin, w, co, DD, HIN, WIN, zd, yin0, xin0, a, bf);
  else
    smallk_frags<TIN, CIN, COUT, TAPS, IS3D, 1>(bin, w, co, DD, HIN, WIN, zd, yin0, xin0, a, bf);

  v8f acc = {};
  acc = __builtin_amdgcn_wmma_f32_16x16x32_f16(false, a, false, bf,
                                               (short)0, acc, false, false);

  if (active && co < COUT) {
    const float sc = HAS_BN ? scale[co] : 1.f;
    const float bi = HAS_BN ? bias[co]  : 0.f;
    const int pixBase = ((b * DD + zd) * HOUT + y) * WOUT + x0 + half * 8;
#pragma unroll
    for (int r = 0; r < 8; ++r) {
      float v = acc[r];
      if (HAS_BN)   v = v * sc + bi;
      if (HAS_RELU) v = fmaxf(v, 0.f);
      out16[(pixBase + r) * COUT + co] = (half_t)v;
    }
  }
}

// ---------------------------------------------------------------------------
// Correlation: cost[b,d,y,x] = mean_c fl[b,y,x,c]*fr[b,y,x-d,c]  (NHWC, C=32)
// ---------------------------------------------------------------------------
__global__ void __launch_bounds__(256)
corr_kernel(const half_t* __restrict__ fl, const half_t* __restrict__ fr,
            half_t* __restrict__ cost, int NB, int DD, int HH, int WW) {
  const int total = NB * DD * HH * WW;
  const int i = blockIdx.x * blockDim.x + threadIdx.x;
  if (i >= total) return;
  const int x = i % WW;  int t = i / WW;
  const int y = t % HH;  t /= HH;
  const int d = t % DD;
  const int b = t / DD;
  float acc = 0.f;
  const int xr = x - d;
  if (xr >= 0) {
    const v8h* pl = (const v8h*)(fl + ((b * HH + y) * WW + x)  * 32);
    const v8h* pr = (const v8h*)(fr + ((b * HH + y) * WW + xr) * 32);
#pragma unroll
    for (int j = 0; j < 4; ++j) {
      const v8h av = pl[j];
      const v8h bv = pr[j];
#pragma unroll
      for (int q = 0; q < 8; ++q) acc += (float)av[q] * (float)bv[q];
    }
  }
  cost[i] = (half_t)(acc * (1.f / 32.f));
}

// ---------------------------------------------------------------------------
// Softmax(-cost) over D + soft-argmin -> disp_low (f32)
// ---------------------------------------------------------------------------
__global__ void __launch_bounds__(256)
softargmin_kernel(const float* __restrict__ cost3, float* __restrict__ dlow,
                  int NB, int DD, int HH, int WW) {
  const int total = NB * HH * WW;
  const int i = blockIdx.x * blockDim.x + threadIdx.x;
  if (i >= total) return;
  const int x = i % WW;  int t = i / WW;
  const int y = t % HH;
  const int b = t / HH;
  const int ds = HH * WW;
  const float* p = cost3 + (b * DD * HH + y) * WW + x;
  float mx = -3.402823e38f;
  for (int d = 0; d < DD; ++d) mx = fmaxf(mx, -p[d * ds]);
  float sum = 0.f, wsum = 0.f;
  for (int d = 0; d < DD; ++d) {
    const float e = expf(-p[d * ds] - mx);
    sum  += e;
    wsum += e * (float)d;
  }
  dlow[i] = wsum / sum;
}

// ---------------------------------------------------------------------------
// Bilinear upsample x4 (half-pixel centers, edge clamp) then *4.0 -> f16
// ---------------------------------------------------------------------------
__global__ void __launch_bounds__(256)
upsample4_kernel(const float* __restrict__ dlow, half_t* __restrict__ dup,
                 int NB, int HL, int WL, int HO, int WO) {
  const int total = NB * HO * WO;
  const int i = blockIdx.x * blockDim.x + threadIdx.x;
  if (i >= total) return;
  const int x = i % WO;  int t = i / WO;
  const int y = t % HO;
  const int b = t / HO;
  const float fy = ((float)y + 0.5f) * 0.25f - 0.5f;
  const float fx = ((float)x + 0.5f) * 0.25f - 0.5f;
  const int y0 = (int)floorf(fy);
  const int x0 = (int)floorf(fx);
  const float wy = fy - (float)y0;
  const float wx = fx - (float)x0;
  const int y0c = iclamp(y0, 0, HL - 1);
  const int y1c = iclamp(y0 + 1, 0, HL - 1);
  const int x0c = iclamp(x0, 0, WL - 1);
  const int x1c = iclamp(x0 + 1, 0, WL - 1);
  const float* p = dlow + b * HL * WL;
  const float v00 = p[y0c * WL + x0c];
  const float v01 = p[y0c * WL + x1c];
  const float v10 = p[y1c * WL + x0c];
  const float v11 = p[y1c * WL + x1c];
  const float v = (1.f - wy) * ((1.f - wx) * v00 + wx * v01)
                +        wy  * ((1.f - wx) * v10 + wx * v11);
  dup[i] = (half_t)(v * 4.0f);
}

// ---------------------------------------------------------------------------
// Host orchestration
// ---------------------------------------------------------------------------
static inline int tileBlocks(long tiles) { return (int)((tiles + 7) / 8); }  // 8 waves / block
static inline int elemBlocks(long n)     { return (int)((n + 255) / 256); }

extern "C" void kernel_launch(void* const* d_in, const int* in_sizes, int n_in,
                              void* d_out, int out_size, void* d_ws, size_t ws_size,
                              hipStream_t stream) {
  (void)in_sizes; (void)n_in; (void)out_size; (void)ws_size;

  const float* img_l = (const float*)d_in[0];
  const float* img_r = (const float*)d_in[1];
  const float* e_w1  = (const float*)d_in[2];
  const float* e_w2  = (const float*)d_in[3];
  const float* e_w3  = (const float*)d_in[4];
  const float* e_w4  = (const float*)d_in[5];
  const float* e_s1  = (const float*)d_in[6];
  const float* e_s2  = (const float*)d_in[7];
  const float* e_s3  = (const float*)d_in[8];
  const float* e_s4  = (const float*)d_in[9];
  const float* e_b1  = (const float*)d_in[10];
  const float* e_b2  = (const float*)d_in[11];
  const float* e_b3  = (const float*)d_in[12];
  const float* e_b4  = (const float*)d_in[13];
  const float* a_w1  = (const float*)d_in[14];
  const float* a_s1  = (const float*)d_in[15];
  const float* a_b1  = (const float*)d_in[16];
  const float* a_w2  = (const float*)d_in[17];
  const float* a_s2  = (const float*)d_in[18];
  const float* a_b2  = (const float*)d_in[19];
  const float* a_w3  = (const float*)d_in[20];
  const float* r_w1  = (const float*)d_in[21];
  const float* r_s1  = (const float*)d_in[22];
  const float* r_b1  = (const float*)d_in[23];
  const float* r_w2  = (const float*)d_in[24];

  const int B = 4, H = 384, W = 768;
  const int H2 = 192, W2 = 384;
  const int H4 = 96,  W4 = 192;
  const int DISP = 48;
  const long vox = (long)DISP * H4 * W4;

  // ---- workspace layout (bytes; all sizes 256B-aligned); NHWC f16 buffers ----
  char* ws = (char*)d_ws;
  const size_t OFF_E1   = 0;                    // (4,192,384,16) f16 = 9437184
  const size_t OFF_E2   = OFF_E1 + 9437184;
  const size_t OFF_E3   = OFF_E2 + 9437184;     // (4,96,192,32) f16 = 4718592
  const size_t OFF_FL   = OFF_E3 + 4718592;
  const size_t OFF_FR   = OFF_FL + 4718592;
  const size_t OFF_C16  = OFF_FR + 4718592;     // (4,48,96,192) f16 = 7077888
  const size_t OFF_A1   = OFF_C16 + 7077888;    // per-batch (48,96,192,8)  = 14155776
  const size_t OFF_A2   = OFF_A1 + 14155776;    // per-batch (48,96,192,16) = 28311552
  const size_t OFF_C3   = OFF_A2 + 28311552;    // (4,48,96,192) f32 = 14155776
  const size_t OFF_DLOW = OFF_C3 + 14155776;    // (4,96,192) f32 = 294912
  const size_t OFF_DUP  = OFF_DLOW + 294912;    // (4,384,768) f16 = 2359296
  const size_t OFF_R1   = OFF_DUP + 2359296;    // (4,384,768,16) f16 = 37748736

  half_t* e1   = (half_t*)(ws + OFF_E1);
  half_t* e2   = (half_t*)(ws + OFF_E2);
  half_t* e3   = (half_t*)(ws + OFF_E3);
  half_t* fl16 = (half_t*)(ws + OFF_FL);
  half_t* fr16 = (half_t*)(ws + OFF_FR);
  half_t* c16  = (half_t*)(ws + OFF_C16);
  half_t* a1   = (half_t*)(ws + OFF_A1);
  half_t* a2   = (half_t*)(ws + OFF_A2);
  float*  c3   = (float*)(ws + OFF_C3);
  float*  dlow = (float*)(ws + OFF_DLOW);
  half_t* dup  = (half_t*)(ws + OFF_DUP);
  half_t* r1   = (half_t*)(ws + OFF_R1);

  float* out_disp = (float*)d_out;                     // (4,1,384,768)
  float* out_fl   = (float*)d_out + (long)B * H * W;   // (4,32,96,192) NCHW

  const long tE12 = (long)B * 1 * H2 * (W2 >> 4);
  const long tE34 = (long)B * 2 * H4 * (W4 >> 4);

  // ---- encoder left ----
  conv_smallk_wmma<float, 3, 16, 9, 2, false, true, true>
      <<<tileBlocks(tE12), 256, 0, stream>>>(img_l, e_w1, e_s1, e_b1, e1,
                                             B, 1, H, W, H2, W2);
  conv2d_nhwc_wmma<16, 16, 1, true, true, 0><<<tileBlocks(tE12), 256, 0, stream>>>(
      e1, e_w2, e_s2, e_b2, e2, nullptr, B, H2, W2, H2, W2);
  conv2d_nhwc_wmma<16, 32, 2, true, true, 0><<<tileBlocks(tE34), 256, 0, stream>>>(
      e2, e_w3, e_s3, e_b3, e3, nullptr, B, H2, W2, H4, W4);
  conv2d_nhwc_wmma<32, 32, 1, true, true, 1><<<tileBlocks(tE34), 256, 0, stream>>>(
      e3, e_w4, e_s4, e_b4, fl16, out_fl, B, H4, W4, H4, W4);  // fl f32 (NCHW)
  // ---- encoder right ----
  conv_smallk_wmma<float, 3, 16, 9, 2, false, true, true>
      <<<tileBlocks(tE12), 256, 0, stream>>>(img_r, e_w1, e_s1, e_b1, e1,
                                             B, 1, H, W, H2, W2);
  conv2d_nhwc_wmma<16, 16, 1, true, true, 0><<<tileBlocks(tE12), 256, 0, stream>>>(
      e1, e_w2, e_s2, e_b2, e2, nullptr, B, H2, W2, H2, W2);
  conv2d_nhwc_wmma<16, 32, 2, true, true, 0><<<tileBlocks(tE34), 256, 0, stream>>>(
      e2, e_w3, e_s3, e_b3, e3, nullptr, B, H2, W2, H4, W4);
  conv2d_nhwc_wmma<32, 32, 1, true, true, 0><<<tileBlocks(tE34), 256, 0, stream>>>(
      e3, e_w4, e_s4, e_b4, fr16, nullptr, B, H4, W4, H4, W4);

  // ---- cost volume ----
  corr_kernel<<<elemBlocks((long)B * vox), 256, 0, stream>>>(
      fl16, fr16, c16, B, DISP, H4, W4);

  // ---- 3D aggregation (per batch) ----
  const long t3d = (long)DISP * H4 * (W4 >> 4);   // NT=1 for all three layers
  for (int b = 0; b < B; ++b) {
    conv_smallk_wmma<half_t, 1, 8, 27, 1, true, true, true>
        <<<tileBlocks(t3d), 256, 0, stream>>>(c16 + (long)b * vox, a_w1, a_s1, a_b1,
                                              a1, 1, DISP, H4, W4, H4, W4);
    conv3d_nhwc_wmma< 8, 16, true,  true,  0><<<tileBlocks(t3d), 256, 0, stream>>>(
        a1, a_w2, a_s2, a_b2, a2, nullptr, DISP, H4, W4);
    conv3d_nhwc_wmma<16,  1, false, false, 2><<<tileBlocks(t3d), 256, 0, stream>>>(
        a2, a_w3, nullptr, nullptr, nullptr, c3 + (long)b * vox, DISP, H4, W4);
  }

  // ---- softmax(-cost) + soft-argmin ----
  softargmin_kernel<<<elemBlocks((long)B * H4 * W4), 256, 0, stream>>>(
      c3, dlow, B, DISP, H4, W4);

  // ---- bilinear x4 upsample ----
  upsample4_kernel<<<elemBlocks((long)B * H * W), 256, 0, stream>>>(
      dlow, dup, B, H4, W4, H, W);

  // ---- refinement head ----
  const long tR = (long)B * 1 * H * (W >> 4);
  conv_smallk_wmma<half_t, 1, 16, 9, 1, false, true, true>
      <<<tileBlocks(tR), 256, 0, stream>>>(dup, r_w1, r_s1, r_b1, r1,
                                           B, 1, H, W, H, W);
  conv2d_nhwc_wmma<16, 1, 1, false, true, 2><<<tileBlocks(tR), 256, 0, stream>>>(
      r1, r_w2, nullptr, nullptr, nullptr, out_disp, B, H, W, H, W);
}
